// DFMNET_34797825032612
// MI455X (gfx1250) — matrix operand order repaired and checked by reading, not compiled, not run on hardware
//
#include <hip/hip_runtime.h>
#include <hip/hip_bf16.h>

typedef __bf16 v16bf __attribute__((ext_vector_type(16)));
typedef __bf16 v8bf  __attribute__((ext_vector_type(8)));
typedef float  v8f   __attribute__((ext_vector_type(8)));

#define B_  2048
#define T_  256
#define I_  64
#define H_  128
#define G_  512     // 4*H gate width
#define K1_ 192     // I + H  (LSTM1 concat input)
#define K2_ 256     // H + H  (LSTM2 concat input)
#define BT_ 16      // batch rows per workgroup
#define KK_ 128
#define O_  64

// workspace byte offsets
#define WS_W1CAT 0          // bf16 [512][192]
#define WS_W2CAT 196608     // bf16 [512][256]
#define WS_B1    458752     // f32  [512]
#define WS_B2    460800     // f32  [512]
#define WS_WK0   462848     // bf16 [128][192]
#define WS_WK1   512000     // bf16 [128][128]
#define WS_WK2   544768
#define WS_WK3   577536
#define WS_WK4   610304
#define WS_WK5   643072     // bf16 [64][128]

__device__ __forceinline__ v16bf cat8(v8bf lo, v8bf hi) {
  union { v16bf v; v8bf h[2]; } u;
  u.h[0] = lo; u.h[1] = hi; return u.v;
}
__device__ __forceinline__ float sigmoidf_(float x) { return 1.0f / (1.0f + __expf(-x)); }
__device__ __forceinline__ float tanhf_(float x) {
  x = fminf(15.0f, fmaxf(-15.0f, x));
  float e = __expf(-2.0f * x);
  return (1.0f - e) / (1.0f + e);
}

// ---------------- weight prep: fp32 -> bf16, concat, fuse biases ----------------
__global__ void prep_lstm_k(const float* __restrict__ Wih1, const float* __restrict__ Whh1,
                            const float* __restrict__ bih1, const float* __restrict__ bhh1,
                            const float* __restrict__ Wih2, const float* __restrict__ Whh2,
                            const float* __restrict__ bih2, const float* __restrict__ bhh2,
                            __bf16* __restrict__ W1cat, __bf16* __restrict__ W2cat,
                            float* __restrict__ b1, float* __restrict__ b2) {
  int i = blockIdx.x * blockDim.x + threadIdx.x;
  const int n1 = G_ * K1_;          // 98304
  const int n2 = G_ * K2_;          // 131072
  if (i < n1) {
    int n = i / K1_, k = i % K1_;
    float v = (k < I_) ? Wih1[n * I_ + k] : Whh1[n * H_ + (k - I_)];
    W1cat[i] = (__bf16)v;
  } else if (i < n1 + n2) {
    int j = i - n1; int n = j / K2_, k = j % K2_;
    float v = (k < H_) ? Wih2[n * H_ + k] : Whh2[n * H_ + (k - H_)];
    W2cat[j] = (__bf16)v;
  } else if (i < n1 + n2 + G_) {
    int n = i - n1 - n2; b1[n] = bih1[n] + bhh1[n];
  } else if (i < n1 + n2 + 2 * G_) {
    int n = i - n1 - n2 - G_; b2[n] = bih2[n] + bhh2[n];
  }
}

__global__ void prep_kdn_k(const float* __restrict__ w0, const float* __restrict__ w1,
                           const float* __restrict__ w2, const float* __restrict__ w3,
                           const float* __restrict__ w4, const float* __restrict__ w5,
                           __bf16* __restrict__ o0, __bf16* __restrict__ o1,
                           __bf16* __restrict__ o2, __bf16* __restrict__ o3,
                           __bf16* __restrict__ o4, __bf16* __restrict__ o5) {
  int i = blockIdx.x * blockDim.x + threadIdx.x;
  const int s0 = KK_ * K1_;   // 24576
  const int s  = KK_ * KK_;   // 16384
  const int s5 = O_ * KK_;    // 8192
  if (i < s0)                      o0[i] = (__bf16)w0[i];
  else if (i < s0 + s)             o1[i - s0] = (__bf16)w1[i - s0];
  else if (i < s0 + 2 * s)         o2[i - s0 - s] = (__bf16)w2[i - s0 - s];
  else if (i < s0 + 3 * s)         o3[i - s0 - 2 * s] = (__bf16)w3[i - s0 - 2 * s];
  else if (i < s0 + 4 * s)         o4[i - s0 - 3 * s] = (__bf16)w4[i - s0 - 3 * s];
  else if (i < s0 + 4 * s + s5)    o5[i - s0 - 4 * s] = (__bf16)w5[i - s0 - 4 * s];
}

// ---------------- fused 2-layer LSTM, persistent over T ----------------
// grid = B/16 workgroups; 512 threads = 16 waves; wave w owns gate columns [32w,32w+32)
__global__ __launch_bounds__(512, 1)
void lstm_fused_k(const float* __restrict__ x,
                  const __bf16* __restrict__ W1cat,
                  const __bf16* __restrict__ W2cat,
                  const float* __restrict__ bias1,
                  const float* __restrict__ bias2,
                  float* __restrict__ outH2,    // [B][128]
                  float* __restrict__ outR) {   // [B][192] = [h2 | x_last]
  __shared__ __bf16 sW1[G_ * K1_];     // 192 KB
  __shared__ float  sGates[BT_ * G_];  //  32 KB
  __shared__ float  sC1[BT_ * H_];     //   8 KB
  __shared__ float  sC2[BT_ * H_];     //   8 KB
  __shared__ float  sH2[BT_ * H_];     //   8 KB
  __shared__ __bf16 sA1[BT_ * K1_];    //   6 KB  [x_t | h1]
  __shared__ __bf16 sA2[BT_ * K2_];    //   8 KB  [h1 | h2]

  const int tid   = threadIdx.x;
  const int lane  = tid & 31;
  const int wave  = tid >> 5;
  const int b0    = blockIdx.x * BT_;
  const int nbase = wave * 32;
  const int lrow  = lane & 15;
  const int lhi   = lane >> 4;

  // cooperative LDS fill of W1cat (16B vectors)
  {
    const uint4* src = (const uint4*)W1cat;
    uint4* dst = (uint4*)sW1;
    const int nvec = G_ * K1_ * 2 / 16;   // 12288
    for (int i = tid; i < nvec; i += 512) dst[i] = src[i];
  }
  for (int i = tid; i < BT_ * H_;  i += 512) { sC1[i] = 0.f; sC2[i] = 0.f; sH2[i] = 0.f; }
  for (int i = tid; i < BT_ * K1_; i += 512) sA1[i] = (__bf16)0.f;
  for (int i = tid; i < BT_ * K2_; i += 512) sA2[i] = (__bf16)0.f;

  // register-cache LSTM2 B-fragments: 2 N-tiles x 8 K-tiles = 128 VGPRs/wave
  v16bf w2f[16];
#pragma unroll
  for (int nt = 0; nt < 2; ++nt)
#pragma unroll
    for (int kt = 0; kt < 8; ++kt) {
      const __bf16* p = W2cat + (size_t)(nbase + nt * 16 + lrow) * K2_ + kt * 32 + (lhi << 4);
      w2f[nt * 8 + kt] = *(const v16bf*)p;
    }

  const float b1a = bias1[nbase + lrow];
  const float b1b = bias1[nbase + 16 + lrow];
  const float b2a = bias2[nbase + lrow];
  const float b2b = bias2[nbase + 16 + lrow];

  __syncthreads();

  for (int t = 0; t < T_; ++t) {
    // stage x_t tile fp32->bf16 into sA1[:, 0:64]; prefetch next step's x
    for (int i = tid; i < BT_ * I_; i += 512) {
      int r = i >> 6, c = i & 63;
      const float* px = &x[(size_t)(b0 + r) * T_ * I_ + (size_t)t * I_ + c];
      sA1[r * K1_ + c] = (__bf16)(*px);
      if (t + 1 < T_) __builtin_prefetch(px + I_, 0, 0);
    }
    __syncthreads();

    // gates1 = [x_t | h1] @ W1cat^T   (A from LDS, B from LDS)
    {
      v8f a0 = {}; v8f a1 = {};
#pragma unroll
      for (int kt = 0; kt < K1_ / 32; ++kt) {
        const __bf16* ap = &sA1[lrow * K1_ + kt * 32 + (lhi << 4)];
        v16bf afrag = cat8(*(const v8bf*)ap, *(const v8bf*)(ap + 8));
        const __bf16* bp0 = &sW1[(nbase + lrow) * K1_ + kt * 32 + (lhi << 4)];
        const __bf16* bp1 = &sW1[(nbase + 16 + lrow) * K1_ + kt * 32 + (lhi << 4)];
        v16bf bf0 = *(const v16bf*)bp0;
        v16bf bf1 = *(const v16bf*)bp1;
        a0 = __builtin_amdgcn_wmma_f32_16x16x32_bf16(false, afrag, false, bf0, (short)0, a0, false, false);
        a1 = __builtin_amdgcn_wmma_f32_16x16x32_bf16(false, afrag, false, bf1, (short)0, a1, false, false);
      }
#pragma unroll
      for (int r = 0; r < 8; ++r) {
        int row = r + (lhi << 3);
        sGates[row * G_ + nbase + lrow]      = a0[r] + b1a;
        sGates[row * G_ + nbase + 16 + lrow] = a1[r] + b1b;
      }
    }
    __syncthreads();

    // elementwise LSTM1 update
    for (int i = tid; i < BT_ * H_; i += 512) {
      int row = i >> 7, j = i & 127;
      const float* g = &sGates[row * G_];
      float ig = sigmoidf_(g[j]);
      float fg = sigmoidf_(g[128 + j]);
      float gg = tanhf_(g[256 + j]);
      float og = sigmoidf_(g[384 + j]);
      float c  = fg * sC1[i] + ig * gg;
      sC1[i] = c;
      float h  = og * tanhf_(c);
      __bf16 hb = (__bf16)h;
      sA1[row * K1_ + I_ + j] = hb;
      sA2[row * K2_ + j]      = hb;
    }
    __syncthreads();

    // gates2 = [h1 | h2] @ W2cat^T   (A from LDS, B from registers)
    {
      v8f a0 = {}; v8f a1 = {};
#pragma unroll
      for (int kt = 0; kt < K2_ / 32; ++kt) {
        const __bf16* ap = &sA2[lrow * K2_ + kt * 32 + (lhi << 4)];
        v16bf afrag = cat8(*(const v8bf*)ap, *(const v8bf*)(ap + 8));
        a0 = __builtin_amdgcn_wmma_f32_16x16x32_bf16(false, afrag, false, w2f[kt],     (short)0, a0, false, false);
        a1 = __builtin_amdgcn_wmma_f32_16x16x32_bf16(false, afrag, false, w2f[8 + kt], (short)0, a1, false, false);
      }
#pragma unroll
      for (int r = 0; r < 8; ++r) {
        int row = r + (lhi << 3);
        sGates[row * G_ + nbase + lrow]      = a0[r] + b2a;
        sGates[row * G_ + nbase + 16 + lrow] = a1[r] + b2b;
      }
    }
    __syncthreads();

    // elementwise LSTM2 update
    for (int i = tid; i < BT_ * H_; i += 512) {
      int row = i >> 7, j = i & 127;
      const float* g = &sGates[row * G_];
      float ig = sigmoidf_(g[j]);
      float fg = sigmoidf_(g[128 + j]);
      float gg = tanhf_(g[256 + j]);
      float og = sigmoidf_(g[384 + j]);
      float c  = fg * sC2[i] + ig * gg;
      sC2[i] = c;
      float h  = og * tanhf_(c);
      sH2[i] = h;
      sA2[row * K2_ + H_ + j] = (__bf16)h;
    }
    __syncthreads();
  }

  // outputs: h2 and r = [h2 | x_last]
  for (int i = tid; i < BT_ * H_; i += 512) {
    int row = i >> 7, j = i & 127;
    float h = sH2[i];
    outH2[(size_t)(b0 + row) * H_ + j] = h;
    outR [(size_t)(b0 + row) * K1_ + j] = h;
  }
  for (int i = tid; i < BT_ * I_; i += 512) {
    int row = i >> 6, c = i & 63;
    outR[(size_t)(b0 + row) * K1_ + H_ + c] =
        x[(size_t)(b0 + row) * T_ * I_ + (size_t)(T_ - 1) * I_ + c];
  }
}

// ---------------- 6-layer KDN head (WMMA, weights from L2) ----------------
// One layer: out[16 x ndim] = relu(in[16 x KD] @ W^T + b); W row-major [ndim][KD] bf16.
// Called by all 256 threads; compute guarded by (nbase < ndim).
__device__ __forceinline__ void kdn_layer(const __bf16* __restrict__ W,
                                          const float* __restrict__ bias,
                                          const __bf16* actIn, int KD,
                                          __bf16* actOut,          // bf16 LDS out (stride KK_), or null
                                          float* yOut, int b0,     // f32 global out (stride O_), or null
                                          int ndim, int lane, int nbase) {
  const int lrow = lane & 15;
  const int lhi  = lane >> 4;
  if (nbase < ndim) {
    v8f acc = {};
    for (int kt = 0; kt < KD / 32; ++kt) {
      const __bf16* ap = actIn + lrow * KD + kt * 32 + (lhi << 4);
      v16bf afrag = cat8(*(const v8bf*)ap, *(const v8bf*)(ap + 8));
      const __bf16* bp = W + (size_t)(nbase + lrow) * KD + kt * 32 + (lhi << 4);
      v16bf bfrag = *(const v16bf*)bp;
      acc = __builtin_amdgcn_wmma_f32_16x16x32_bf16(false, afrag, false, bfrag, (short)0, acc, false, false);
    }
    float bv = bias[nbase + lrow];
    if (actOut) {
#pragma unroll
      for (int r = 0; r < 8; ++r) {
        int row = r + (lhi << 3);
        float v = acc[r] + bv;
        v = v > 0.f ? v : 0.f;
        actOut[row * KK_ + nbase + lrow] = (__bf16)v;
      }
    } else {
#pragma unroll
      for (int r = 0; r < 8; ++r) {
        int row = r + (lhi << 3);
        yOut[(size_t)(b0 + row) * O_ + nbase + lrow] = acc[r] + bv;
      }
    }
  }
}

// grid = B/16; 256 threads = 8 waves; wave w owns 16 output columns
__global__ __launch_bounds__(256, 1)
void kdn_fused_k(const float* __restrict__ rIn,   // [B][192]
                 const __bf16* __restrict__ wk0, const __bf16* __restrict__ wk1,
                 const __bf16* __restrict__ wk2, const __bf16* __restrict__ wk3,
                 const __bf16* __restrict__ wk4, const __bf16* __restrict__ wk5,
                 const float* __restrict__ bk0, const float* __restrict__ bk1,
                 const float* __restrict__ bk2, const float* __restrict__ bk3,
                 const float* __restrict__ bk4, const float* __restrict__ bk5,
                 float* __restrict__ y) {          // [B][64]
  __shared__ __bf16 s0[BT_ * K1_];
  __shared__ __bf16 s1[BT_ * KK_];
  __shared__ __bf16 s2[BT_ * KK_];

  const int tid   = threadIdx.x;
  const int lane  = tid & 31;
  const int wave  = tid >> 5;
  const int b0    = blockIdx.x * BT_;
  const int nbase = wave * 16;

  for (int i = tid; i < BT_ * K1_; i += 256)
    s0[i] = (__bf16)rIn[(size_t)(b0 + (i / K1_)) * K1_ + (i % K1_)];
  __syncthreads();

  kdn_layer(wk0, bk0, s0, K1_, s1, (float*)0, b0, KK_, lane, nbase); __syncthreads();
  kdn_layer(wk1, bk1, s1, KK_, s2, (float*)0, b0, KK_, lane, nbase); __syncthreads();
  kdn_layer(wk2, bk2, s2, KK_, s1, (float*)0, b0, KK_, lane, nbase); __syncthreads();
  kdn_layer(wk3, bk3, s1, KK_, s2, (float*)0, b0, KK_, lane, nbase); __syncthreads();
  kdn_layer(wk4, bk4, s2, KK_, s1, (float*)0, b0, KK_, lane, nbase); __syncthreads();
  kdn_layer(wk5, bk5, s1, KK_, (__bf16*)0, y, b0, O_, lane, nbase);
}

extern "C" void kernel_launch(void* const* d_in, const int* in_sizes, int n_in,
                              void* d_out, int out_size, void* d_ws, size_t ws_size,
                              hipStream_t stream) {
  const float* x    = (const float*)d_in[0];
  const float* Wih1 = (const float*)d_in[1];
  const float* Whh1 = (const float*)d_in[2];
  const float* bih1 = (const float*)d_in[3];
  const float* bhh1 = (const float*)d_in[4];
  const float* Wih2 = (const float*)d_in[5];
  const float* Whh2 = (const float*)d_in[6];
  const float* bih2 = (const float*)d_in[7];
  const float* bhh2 = (const float*)d_in[8];
  const float* Wk0 = (const float*)d_in[9];   const float* bk0 = (const float*)d_in[10];
  const float* Wk1 = (const float*)d_in[11];  const float* bk1 = (const float*)d_in[12];
  const float* Wk2 = (const float*)d_in[13];  const float* bk2 = (const float*)d_in[14];
  const float* Wk3 = (const float*)d_in[15];  const float* bk3 = (const float*)d_in[16];
  const float* Wk4 = (const float*)d_in[17];  const float* bk4 = (const float*)d_in[18];
  const float* Wk5 = (const float*)d_in[19];  const float* bk5 = (const float*)d_in[20];

  char* ws = (char*)d_ws;
  __bf16* W1cat = (__bf16*)(ws + WS_W1CAT);
  __bf16* W2cat = (__bf16*)(ws + WS_W2CAT);
  float*  b1    = (float*)(ws + WS_B1);
  float*  b2    = (float*)(ws + WS_B2);
  __bf16* wk0 = (__bf16*)(ws + WS_WK0);
  __bf16* wk1 = (__bf16*)(ws + WS_WK1);
  __bf16* wk2 = (__bf16*)(ws + WS_WK2);
  __bf16* wk3 = (__bf16*)(ws + WS_WK3);
  __bf16* wk4 = (__bf16*)(ws + WS_WK4);
  __bf16* wk5 = (__bf16*)(ws + WS_WK5);

  float* y     = (float*)d_out;                       // [B][64]
  float* outH2 = (float*)d_out + (size_t)B_ * O_;     // [B][128]
  float* outR  = outH2 + (size_t)B_ * H_;             // [B][192]

  {
    int n = G_ * K1_ + G_ * K2_ + 2 * G_;
    prep_lstm_k<<<(n + 255) / 256, 256, 0, stream>>>(Wih1, Whh1, bih1, bhh1,
                                                     Wih2, Whh2, bih2, bhh2,
                                                     W1cat, W2cat, b1, b2);
  }
  {
    int n = KK_ * K1_ + 4 * KK_ * KK_ + O_ * KK_;
    prep_kdn_k<<<(n + 255) / 256, 256, 0, stream>>>(Wk0, Wk1, Wk2, Wk3, Wk4, Wk5,
                                                    wk0, wk1, wk2, wk3, wk4, wk5);
  }
  lstm_fused_k<<<B_ / BT_, 512, 0, stream>>>(x, W1cat, W2cat, b1, b2, outH2, outR);
  kdn_fused_k<<<B_ / BT_, 256, 0, stream>>>(outR, wk0, wk1, wk2, wk3, wk4, wk5,
                                            bk0, bk1, bk2, bk3, bk4, bk5, y);
}
